// Trader_24962349924359
// MI455X (gfx1250) — matrix operand order repaired and checked by reading, not compiled, not run on hardware
//
#include <hip/hip_runtime.h>
#include <hip/hip_bf16.h>
#include <math.h>

// ---------------------------------------------------------------------------
// Trader forward pass for gfx1250 (MI455X).
// GEMMs: v_wmma_f32_16x16x32_f16, fragment-swizzled f16 weights (2x b128 per
// B fragment), double-buffered async global->LDS A-tile staging (ASYNCcnt),
// 2x ds_load_b128 per A fragment, global prefetch one K-tile ahead,
// native-rcp activations.
// ---------------------------------------------------------------------------

#define B_   2048
#define T_   128
#define CAT_ 8
#define DAYS_ 32
#define FEATS_ 5
#define A_   128
#define KCAT_ 2320          // 16 + 128 + 16*128 + 128
#define KCATP_ 2336         // padded to multiple of 32
#define NZ_  8128           // A*(A-1)/2

typedef __attribute__((ext_vector_type(16))) _Float16 v16h;
typedef __attribute__((ext_vector_type(8)))  float    v8f;

__device__ __forceinline__ float silu_f(float v) {
    // v * sigmoid(v) with native v_rcp_f32 (activation: exact IEEE div unneeded)
    return v * __builtin_amdgcn_rcpf(1.0f + __expf(-v));
}

__device__ __forceinline__ float tanh_f(float v) {
    // tanh(v) = 1 - 2/(exp(2v)+1), native exp + rcp
    return 1.0f - 2.0f * __builtin_amdgcn_rcpf(__expf(2.0f * v) + 1.0f);
}

// A-fragment K index, this code's contiguous order k = 16*hs + h (lane half hs, half-slot h), same as B packing
__device__ __forceinline__ int a_k_of(int h, int hs) {
    return hs * 16 + h;
}

// gfx1250 async global->LDS 4-byte lane copy (ASYNCcnt path)
__device__ __forceinline__ void async_copy_b32(_Float16* ldsDst, const _Float16* gsrc) {
    unsigned lds_addr = (unsigned)(size_t)ldsDst;
    unsigned long long gaddr = (unsigned long long)(size_t)gsrc;
    asm volatile("global_load_async_to_lds_b32 %0, %1, off"
                 :: "v"(lds_addr), "v"(gaddr)
                 : "memory");
}

__device__ __forceinline__ void wait_async0() {
#if __has_builtin(__builtin_amdgcn_s_wait_asynccnt)
    __builtin_amdgcn_s_wait_asynccnt(0);
#else
    asm volatile("s_wait_asynccnt 0" ::: "memory");
#endif
}

// ---------------------------------------------------------------------------
// weight conversion f32 -> f16, fragment-swizzled.  grid = (ktiles, ntiles),
// one 512-half tile per 256-thread block (no integer division anywhere).
// tile (nt, kt) lives at ((nt*ktiles)+kt)*512 halves; lane L's fragment
// halves h=0..15 are contiguous at lane*16+h holding
// W[kt*32 + (L>>4)*16 + h][nt*16 + (L&15)].  Rows >= K are zero (K padding).
// ---------------------------------------------------------------------------
__global__ void cvt_swz_kernel(const float* __restrict__ s, _Float16* __restrict__ d,
                               int K, int N) {
    int kt = blockIdx.x, nt = blockIdx.y, ktiles = gridDim.x;
    int e    = threadIdx.x * 2;        // element pair within tile
    int lane = e >> 4;
    int h0   = e & 15;
    int k = kt * 32 + (lane >> 4) * 16 + h0;
    int n = nt * 16 + (lane & 15);
    size_t base = ((size_t)nt * ktiles + kt) * 512;
    d[base + e]     = (k     < K) ? (_Float16)s[(size_t)k * N + n]       : (_Float16)0.0f;
    d[base + e + 1] = (k + 1 < K) ? (_Float16)s[(size_t)(k + 1) * N + n] : (_Float16)0.0f;
}

// ---------------------------------------------------------------------------
// arranger: per batch, perf-rank tickers (descending), gather e, p, x
// ---------------------------------------------------------------------------
__global__ void arrange_kernel(const float* __restrict__ x, const float* __restrict__ e,
                               const float* __restrict__ y, const float* __restrict__ p,
                               int* __restrict__ orders, float* __restrict__ ea,
                               float* __restrict__ pa, _Float16* __restrict__ xa16) {
    int bb = blockIdx.x;
    int t  = threadIdx.x;

    const float* yrow = y + (((size_t)bb * T_ + t) * DAYS_) * FEATS_ + 1; // CLOSE_IDX=1
    int first = 0;
    for (int d = DAYS_ - 1; d >= 0; --d)
        if (yrow[(size_t)d * FEATS_] != 0.0f) first = d;
    float start = yrow[(size_t)first * FEATS_];
    float last  = yrow[(size_t)(DAYS_ - 1) * FEATS_];
    float perf  = (start == 0.0f) ? 0.0f : (last - start) / start;

    __shared__ float key[T_];
    __shared__ int   val[T_];
    key[t] = perf; val[t] = t;
    __syncthreads();

    for (int k = 2; k <= T_; k <<= 1) {
        for (int j = k >> 1; j > 0; j >>= 1) {
            int ixj = t ^ j;
            if (ixj > t) {
                float ka = key[t], kb = key[ixj];
                int   va = val[t], vb = val[ixj];
                bool before = (ka > kb) || (ka == kb && va < vb);
                bool up = ((t & k) == 0);
                if (up ? !before : before) {
                    key[t] = kb; key[ixj] = ka;
                    val[t] = vb; val[ixj] = va;
                }
            }
            __syncthreads();
        }
    }

    int ord = val[t];
    orders[(size_t)bb * T_ + t] = ord;
    float ev = e[(size_t)bb * T_ + ord];
    float pv = p[(size_t)bb * T_ + ord];
    ea[(size_t)bb * T_ + t] = ev;
    pa[(size_t)bb * T_ + t] = pv;
    const float* xr = x + ((size_t)bb * T_ + ord) * CAT_;
    _Float16* xo = xa16 + ((size_t)bb * T_ + t) * CAT_;
    #pragma unroll
    for (int cat = 0; cat < CAT_; ++cat) xo[cat] = (_Float16)xr[cat];
}

// ---------------------------------------------------------------------------
// scaled_e / scaled_p into concat + zero K-padding columns
// ---------------------------------------------------------------------------
__global__ void fill_concat_kernel(const float* __restrict__ ea, const float* __restrict__ pa,
                                   const float* __restrict__ c, _Float16* __restrict__ concat) {
    int bb = blockIdx.x, t = threadIdx.x;
    float cc = c[bb];
    float pv = pa[(size_t)bb * T_ + t];
    float ev = ea[(size_t)bb * T_ + t];
    float sp = pv / (cc + 1e-8f);
    _Float16* row = concat + (size_t)bb * KCATP_;
    row[16 + t]                = (_Float16)(sp * ev);          // scaled_e
    row[16 + T_ + 16 * T_ + t] = (_Float16)sp;                 // scaled_p at 2192
    if (t < KCATP_ - KCAT_) row[KCAT_ + t] = (_Float16)0.0f;   // pad cols 2320..2335
}

// ---------------------------------------------------------------------------
// temporal DNN: 16 arranged (b,t) rows: normalize(32x5) -> 160->64->32->16
// block = 128 threads (4 waves); WMMA with swizzled weights (v16h loads)
// ---------------------------------------------------------------------------
__global__ void temporal_kernel(const float* __restrict__ y, const int* __restrict__ orders,
                                const _Float16* __restrict__ W1s, const float* __restrict__ b1,
                                const _Float16* __restrict__ W2s, const float* __restrict__ b2,
                                const _Float16* __restrict__ W3s, const float* __restrict__ b3,
                                _Float16* __restrict__ concat) {
    __shared__ _Float16 Yn[16][168];
    __shared__ _Float16 H1[16][72];
    __shared__ _Float16 H2[16][40];

    int tid  = threadIdx.x;
    int lane = tid & 31;
    int wave = tid >> 5;
    int ar0  = blockIdx.x * 16;
    int bb   = ar0 / T_;

    if (tid < 16 * FEATS_) {
        int r = tid / FEATS_, f = tid % FEATS_;
        int t = (ar0 + r) % T_;
        int src = orders[(size_t)bb * T_ + t];
        const float* yp = y + (((size_t)bb * T_ + src) * DAYS_) * FEATS_ + f;
        float s = 0.0f, s2 = 0.0f;
        #pragma unroll
        for (int d = 0; d < DAYS_; ++d) {
            float v = yp[(size_t)d * FEATS_];
            s += v; s2 += v * v;
        }
        float mean = s * (1.0f / DAYS_);
        float var  = fmaxf(s2 * (1.0f / DAYS_) - mean * mean, 0.0f);
        float inv  = __builtin_amdgcn_rcpf(sqrtf(var) + 1e-8f);
        #pragma unroll
        for (int d = 0; d < DAYS_; ++d)
            Yn[r][d * FEATS_ + f] = (_Float16)((yp[(size_t)d * FEATS_] - mean) * inv);
    }
    __syncthreads();

    int m  = lane & 15;
    int hs = lane >> 4;

    // layer1: 160 -> 64, wave w owns N-tile w (ktiles=5)
    {
        v8f acc = {};
        for (int kt = 0; kt < 5; ++kt) {
            v16h af, bf;
            #pragma unroll
            for (int h = 0; h < 16; ++h) af[h] = Yn[m][kt * 32 + a_k_of(h, hs)];
            bf = *(const v16h*)(W1s + ((size_t)wave * 5 + kt) * 512 + lane * 16);
            acc = __builtin_amdgcn_wmma_f32_16x16x32_f16(false, af, false, bf,
                                                         (short)0, acc, false, false);
        }
        #pragma unroll
        for (int r = 0; r < 8; ++r)
            H1[hs * 8 + r][wave * 16 + m] = (_Float16)silu_f(acc[r] + b1[wave * 16 + m]);
    }
    __syncthreads();

    // layer2: 64 -> 32, waves 0..1 (ktiles=2)
    if (wave < 2) {
        v8f acc = {};
        for (int kt = 0; kt < 2; ++kt) {
            v16h af, bf;
            #pragma unroll
            for (int h = 0; h < 16; ++h) af[h] = H1[m][kt * 32 + a_k_of(h, hs)];
            bf = *(const v16h*)(W2s + ((size_t)wave * 2 + kt) * 512 + lane * 16);
            acc = __builtin_amdgcn_wmma_f32_16x16x32_f16(false, af, false, bf,
                                                         (short)0, acc, false, false);
        }
        #pragma unroll
        for (int r = 0; r < 8; ++r)
            H2[hs * 8 + r][wave * 16 + m] = (_Float16)silu_f(acc[r] + b2[wave * 16 + m]);
    }
    __syncthreads();

    // layer3: 32 -> 16, wave 0 (ktiles=1), write to concat
    if (wave == 0) {
        v8f acc = {};
        v16h af, bf;
        #pragma unroll
        for (int h = 0; h < 16; ++h) af[h] = H2[m][a_k_of(h, hs)];
        bf = *(const v16h*)(W3s + lane * 16);
        acc = __builtin_amdgcn_wmma_f32_16x16x32_f16(false, af, false, bf,
                                                     (short)0, acc, false, false);
        #pragma unroll
        for (int r = 0; r < 8; ++r) {
            int mm = hs * 8 + r;
            int t  = (ar0 + mm) % T_;
            float v = silu_f(acc[r] + b3[m]);
            concat[(size_t)bb * KCATP_ + (16 + T_) + t * 16 + m] = (_Float16)v;
        }
    }
}

// ---------------------------------------------------------------------------
// generic f16 WMMA GEMM:  out = act(A[MxK] * Wswz + bias)
// block = 256 threads (8 waves); wave computes a 16x64 C strip (4 acc tiles).
// A tile (16x32) staged fragment-swizzled in double-buffered LDS via async
// global->LDS copies that overlap the WMMAs of the previous K-tile.
// ---------------------------------------------------------------------------
__global__ void gemm_wmma_kernel(const _Float16* __restrict__ A, int lda,
                                 const _Float16* __restrict__ Wswz,
                                 const float* __restrict__ bias,
                                 _Float16* __restrict__ outH, float* __restrict__ outF,
                                 int ldc, int colOff,
                                 int K, int N, int act) {
    __shared__ _Float16 As[2][512] __attribute__((aligned(32)));

    int tid  = threadIdx.x;
    int lane = tid & 31;
    int wave = tid >> 5;
    int row0 = blockIdx.y * 16;
    int ktiles = K >> 5;
    int ntBase = blockIdx.x * 32 + wave * 4;   // N-tile base for this wave

    // staging source for this thread: 2 contiguous halves of the swizzled A tile
    int e      = tid * 2;                       // swizzled element index (even)
    int lane_t = e >> 4;
    int hh     = e & 15;                        // even -> (k, k+1) contiguous
    int mrow   = lane_t & 15;
    int hsl    = lane_t >> 4;
    int kfrag  = a_k_of(hh, hsl);
    const _Float16* Arow = A + (size_t)(row0 + mrow) * lda + kfrag;

    v8f acc0 = {}, acc1 = {}, acc2 = {}, acc3 = {};

    // prime the pipeline: stage K-tile 0 into buffer 0
    async_copy_b32(&As[0][e], Arow);

    for (int kt = 0; kt < ktiles; ++kt) {
        wait_async0();                 // copy issued last iteration complete
        __syncthreads();               // visible to all waves; prior reads done
        if (kt + 1 < ktiles)           // overlap next A tile with this compute
            async_copy_b32(&As[(kt + 1) & 1][e], Arow + (size_t)(kt + 1) * 32);

        v16h af = *(const v16h*)(&As[kt & 1][lane * 16]);   // 2x ds_load_b128
        #pragma unroll
        for (int s = 0; s < 4; ++s) {
            int nt = ntBase + s;
            if (nt * 16 < N) {
                const _Float16* wp = Wswz + ((size_t)nt * ktiles + kt) * 512 + lane * 16;
                if (kt + 1 < ktiles) __builtin_prefetch(wp + 512, 0, 1);
                v16h bf = *(const v16h*)wp;                 // 2x global_load_b128
                v8f& acc = (s == 0) ? acc0 : (s == 1) ? acc1 : (s == 2) ? acc2 : acc3;
                acc = __builtin_amdgcn_wmma_f32_16x16x32_f16(false, af, false, bf,
                                                             (short)0, acc, false, false);
            }
        }
    }

    int m = lane & 15, hs = lane >> 4;
    #pragma unroll
    for (int s = 0; s < 4; ++s) {
        int nt = ntBase + s;
        int n0 = nt * 16;
        if (n0 < N) {
            v8f acc = (s == 0) ? acc0 : (s == 1) ? acc1 : (s == 2) ? acc2 : acc3;
            #pragma unroll
            for (int r = 0; r < 8; ++r) {
                int mm = hs * 8 + r;
                float v = acc[r] + (bias ? bias[n0 + m] : 0.0f);
                if (act == 1)      v = silu_f(v);
                else if (act == 2) v = tanh_f(v);
                size_t o = (size_t)(row0 + mm) * ldc + colOff + n0 + m;
                if (outH) outH[o] = (_Float16)v;
                if (outF) outF[o] = v;
            }
        }
    }
}

// ---------------------------------------------------------------------------
// critic: value[b] = dot(main[b], crit_W) + crit_b
// ---------------------------------------------------------------------------
__global__ void critic_kernel(const _Float16* __restrict__ mainAct, const float* __restrict__ W,
                              const float* __restrict__ bias, float* __restrict__ out) {
    __shared__ float red[256];
    int bb = blockIdx.x, tid = threadIdx.x;
    float s = 0.0f;
    for (int i = tid; i < 8 * A_; i += 256)
        s += (float)mainAct[(size_t)bb * 8 * A_ + i] * W[i];
    red[tid] = s; __syncthreads();
    for (int off = 128; off > 0; off >>= 1) {
        if (tid < off) red[tid] += red[tid + off];
        __syncthreads();
    }
    if (tid == 0) out[bb] = red[0] + bias[0];
}

// ---------------------------------------------------------------------------
// mu post-processing
// ---------------------------------------------------------------------------
__global__ void postmu_kernel(const float* __restrict__ buy, const float* __restrict__ sell,
                              const float* __restrict__ pa, const float* __restrict__ ea,
                              const float* __restrict__ c, float* __restrict__ outMu) {
    __shared__ float red[A_];
    int bb = blockIdx.x, t = threadIdx.x;
    float pv = pa[(size_t)bb * A_ + t];
    float ev = ea[(size_t)bb * A_ + t];
    float cc = c[bb];
    float bv = buy[(size_t)bb * A_ + t];
    float sv = sell[(size_t)bb * A_ + t];
    float um = (bv > sv) ? bv : -sv;
    float lb = -(pv * ev);
    float mu_cash = um * (cc - lb);
    float mu = (pv == 0.0f) ? 0.0f : mu_cash / pv;

    red[t] = (mu > 0.0f) ? mu * pv : 0.0f;
    __syncthreads();
    for (int off = 64; off > 0; off >>= 1) {
        if (t < off) red[t] += red[t + off];
        __syncthreads();
    }
    float needed = red[0];
    float ratios = needed / (cc + 1e-8f);
    if (mu > 0.0f && ratios >= 0.9f) mu = mu / (ratios * 1.1f + 0.001f);
    outMu[(size_t)bb * A_ + t] = fmaxf(mu, -ev);
}

// ---------------------------------------------------------------------------
// Cholesky from partial correlations; thread i = row i, sequential over cols
// ---------------------------------------------------------------------------
__global__ void cholesky_kernel(const float* __restrict__ zvec, const float* __restrict__ pa,
                                float* __restrict__ Lout) {
    int bb = blockIdx.x, i = threadIdx.x;
    const float* zr = zvec + (size_t)bb * NZ_;
    float pv = pa[(size_t)bb * A_ + i];
    float inv_p = (pv == 0.0f) ? 0.0f : 1.0f / pv;
    float sumsq = 0.0f;
    int zbase = i * (i - 1) / 2;
    float* Lrow = Lout + ((size_t)bb * A_ + i) * A_;
    for (int j = 0; j < A_; ++j) {
        float v = 0.0f;
        if (i > j) {
            float z = zr[zbase + j];
            z = fminf(fmaxf(z, -1.0f + 1e-6f), 1.0f - 1e-6f);
            v = z * sqrtf(fmaxf(1.0f - sumsq, 1e-12f));
        } else if (i == j) {
            v = sqrtf(fmaxf(1.0f - sumsq, 1e-12f));
        }
        sumsq += v * v;
        float lv = v * inv_p;
        if (i == j) lv += 1e-6f;
        Lrow[j] = lv;
    }
}

// ---------------------------------------------------------------------------
// host launcher
// ---------------------------------------------------------------------------
extern "C" void kernel_launch(void* const* d_in, const int* in_sizes, int n_in,
                              void* d_out, int out_size, void* d_ws, size_t ws_size,
                              hipStream_t stream) {
    (void)in_sizes; (void)n_in; (void)out_size; (void)ws_size;

    const float* x  = (const float*)d_in[0];
    const float* e  = (const float*)d_in[1];
    const float* y  = (const float*)d_in[2];
    const float* p  = (const float*)d_in[3];
    const float* c  = (const float*)d_in[4];
    const float* enc_W1 = (const float*)d_in[5];
    const float* enc_b1 = (const float*)d_in[6];
    const float* enc_W2 = (const float*)d_in[7];
    const float* enc_b2 = (const float*)d_in[8];
    const float* tmp_W1 = (const float*)d_in[9];
    const float* tmp_b1 = (const float*)d_in[10];
    const float* tmp_W2 = (const float*)d_in[11];
    const float* tmp_b2 = (const float*)d_in[12];
    const float* tmp_W3 = (const float*)d_in[13];
    const float* tmp_b3 = (const float*)d_in[14];
    const float* com_W1 = (const float*)d_in[15];
    const float* com_b1 = (const float*)d_in[16];
    const float* com_W2 = (const float*)d_in[17];
    const float* com_b2 = (const float*)d_in[18];
    const float* com_W3 = (const float*)d_in[19];
    const float* com_b3 = (const float*)d_in[20];
    const float* crit_W = (const float*)d_in[21];
    const float* crit_b = (const float*)d_in[22];
    const float* act_W1 = (const float*)d_in[23];
    const float* act_b1 = (const float*)d_in[24];
    const float* act_W2 = (const float*)d_in[25];
    const float* act_b2 = (const float*)d_in[26];
    const float* buy_W  = (const float*)d_in[27];
    const float* buy_b  = (const float*)d_in[28];
    const float* sell_W = (const float*)d_in[29];
    const float* sell_b = (const float*)d_in[30];
    const float* z_W    = (const float*)d_in[31];
    const float* z_b    = (const float*)d_in[32];

    float* out = (float*)d_out;
    float* outMu  = out;                                          // (B, A)
    float* outL   = out + (size_t)B_ * A_;                        // (B, A, A)
    float* outVal = out + (size_t)B_ * A_ + (size_t)B_ * A_ * A_; // (B, 1)

    char* ws = (char*)d_ws;
    size_t off = 0;
    auto alloc = [&](size_t bytes) -> char* {
        char* q = ws + off;
        off = (off + bytes + 255) & ~(size_t)255;
        return q;
    };

    int*       wOrd   = (int*)      alloc((size_t)B_ * T_ * 4);
    float*     wEa    = (float*)    alloc((size_t)B_ * T_ * 4);
    float*     wPa    = (float*)    alloc((size_t)B_ * T_ * 4);
    _Float16*  wXa    = (_Float16*) alloc((size_t)B_ * T_ * CAT_ * 2);
    _Float16*  wCat   = (_Float16*) alloc((size_t)B_ * KCATP_ * 2);
    _Float16*  wEnc1  = (_Float16*) alloc((size_t)B_ * 32 * 2);
    _Float16*  wM1    = (_Float16*) alloc((size_t)B_ * 32 * A_ * 2);
    _Float16*  wM2    = (_Float16*) alloc((size_t)B_ * 16 * A_ * 2);
    _Float16*  wMain  = (_Float16*) alloc((size_t)B_ * 8 * A_ * 2);
    _Float16*  wHid   = (_Float16*) alloc((size_t)B_ * 8 * A_ * 2);
    _Float16*  wHid2  = (_Float16*) alloc((size_t)B_ * 4 * A_ * 2);
    float*     wBuy   = (float*)    alloc((size_t)B_ * A_ * 4);
    float*     wSell  = (float*)    alloc((size_t)B_ * A_ * 4);
    float*     wZ     = (float*)    alloc((size_t)B_ * NZ_ * 4);
    _Float16*  hEncW1 = (_Float16*) alloc((size_t)1024 * 32 * 2);
    _Float16*  hEncW2 = (_Float16*) alloc((size_t)32 * 16 * 2);
    _Float16*  hTmpW1 = (_Float16*) alloc((size_t)160 * 64 * 2);
    _Float16*  hTmpW2 = (_Float16*) alloc((size_t)64 * 32 * 2);
    _Float16*  hTmpW3 = (_Float16*) alloc((size_t)32 * 16 * 2);
    _Float16*  hComW1 = (_Float16*) alloc((size_t)KCATP_ * 32 * A_ * 2);
    _Float16*  hComW2 = (_Float16*) alloc((size_t)32 * A_ * 16 * A_ * 2);
    _Float16*  hComW3 = (_Float16*) alloc((size_t)16 * A_ * 8 * A_ * 2);
    _Float16*  hActW1 = (_Float16*) alloc((size_t)8 * A_ * 8 * A_ * 2);
    _Float16*  hActW2 = (_Float16*) alloc((size_t)8 * A_ * 4 * A_ * 2);
    _Float16*  hBuyW  = (_Float16*) alloc((size_t)4 * A_ * A_ * 2);
    _Float16*  hSellW = (_Float16*) alloc((size_t)4 * A_ * A_ * 2);
    _Float16*  hZW    = (_Float16*) alloc((size_t)8 * A_ * NZ_ * 2);

    // ---- fragment-swizzled weight conversion: grid=(ktiles, ntiles) ----
    #define SWZ(src, dst, K, Kpad, N)                                            \
        cvt_swz_kernel<<<dim3((Kpad) / 32, (N) / 16), 256, 0, stream>>>(         \
            src, dst, K, N)
    SWZ(enc_W1, hEncW1, 1024, 1024, 32);
    SWZ(enc_W2, hEncW2, 32,   32,   16);
    SWZ(tmp_W1, hTmpW1, 160,  160,  64);
    SWZ(tmp_W2, hTmpW2, 64,   64,   32);
    SWZ(tmp_W3, hTmpW3, 32,   32,   16);
    SWZ(com_W1, hComW1, KCAT_, KCATP_, 32 * A_);
    SWZ(com_W2, hComW2, 32 * A_, 32 * A_, 16 * A_);
    SWZ(com_W3, hComW3, 16 * A_, 16 * A_, 8 * A_);
    SWZ(act_W1, hActW1, 8 * A_, 8 * A_, 8 * A_);
    SWZ(act_W2, hActW2, 8 * A_, 8 * A_, 4 * A_);
    SWZ(buy_W,  hBuyW,  4 * A_, 4 * A_, A_);
    SWZ(sell_W, hSellW, 4 * A_, 4 * A_, A_);
    SWZ(z_W,    hZW,    8 * A_, 8 * A_, NZ_);
    #undef SWZ

    // ---- arrange + concat scalars ----
    arrange_kernel<<<B_, T_, 0, stream>>>(x, e, y, p, wOrd, wEa, wPa, wXa);
    fill_concat_kernel<<<B_, T_, 0, stream>>>(wEa, wPa, c, wCat);

    // ---- temporal DNN (WMMA) ----
    temporal_kernel<<<(B_ * T_) / 16, 128, 0, stream>>>(
        y, wOrd, hTmpW1, tmp_b1, hTmpW2, tmp_b2, hTmpW3, tmp_b3, wCat);

    // ---- GEMM helper ----
    auto gemm = [&](const _Float16* Ain, int lda, const _Float16* Win, const float* bias,
                    _Float16* oH, float* oF, int ldc, int colOff, int M, int K, int N, int act) {
        dim3 g((N + 511) / 512, M / 16);
        gemm_wmma_kernel<<<g, 256, 0, stream>>>(Ain, lda, Win, bias, oH, oF,
                                                ldc, colOff, K, N, act);
    };

    // encoder: 1024 -> 32 -> 16 (silu), layer2 lands in concat cols [0,16)
    gemm(wXa,   1024, hEncW1, enc_b1, wEnc1, nullptr, 32,     0, B_, 1024, 32, 1);
    gemm(wEnc1, 32,   hEncW2, enc_b2, wCat,  nullptr, KCATP_, 0, B_, 32,   16, 1);

    // common trunk: 2336 -> 4096 -> 2048 -> 1024 (silu)
    gemm(wCat, KCATP_, hComW1, com_b1, wM1,   nullptr, 32 * A_, 0, B_, KCATP_, 32 * A_, 1);
    gemm(wM1,  32*A_,  hComW2, com_b2, wM2,   nullptr, 16 * A_, 0, B_, 32*A_,  16 * A_, 1);
    gemm(wM2,  16*A_,  hComW3, com_b3, wMain, nullptr, 8 * A_,  0, B_, 16*A_,  8 * A_,  1);

    // critic
    critic_kernel<<<B_, 256, 0, stream>>>(wMain, crit_W, crit_b, outVal);

    // actor
    gemm(wMain, 8*A_, hActW1, act_b1, wHid,  nullptr, 8 * A_, 0, B_, 8*A_, 8 * A_, 1);
    gemm(wHid,  8*A_, hActW2, act_b2, wHid2, nullptr, 4 * A_, 0, B_, 8*A_, 4 * A_, 1);
    gemm(wHid2, 4*A_, hBuyW,  buy_b,  nullptr, wBuy,  A_,     0, B_, 4*A_, A_,     1);
    gemm(wHid2, 4*A_, hSellW, sell_b, nullptr, wSell, A_,     0, B_, 4*A_, A_,     1);
    gemm(wHid,  8*A_, hZW,    z_b,    nullptr, wZ,    NZ_,    0, B_, 8*A_, NZ_,    2);

    // mu post-processing + Cholesky
    postmu_kernel<<<B_, A_, 0, stream>>>(wBuy, wSell, wPa, wEa, c, outMu);
    cholesky_kernel<<<B_, A_, 0, stream>>>(wZ, wPa, outL);
}
